// PTMHRMF_35725537968314
// MI455X (gfx1250) — compile-verified, matrix operand order checked by reading, not run on hardware
//
#include <hip/hip_runtime.h>
#include <hip/hip_bf16.h>
#include <math.h>

// ---- problem constants (from reference) ----
#define BB   8
#define SS   512
#define HH   768
#define NHH  12
#define HD   64      // HH/NHH
#define DW_  300
#define DWP  320     // DW padded to multiple of 32
#define WW   256
#define LL   4
#define NLL  2

typedef __bf16 bf16;
typedef __attribute__((ext_vector_type(16))) __bf16 v16bf;
typedef __attribute__((ext_vector_type(8)))  __bf16 v8bf;
typedef __attribute__((ext_vector_type(8)))  float  v8f;

// ---------------------------------------------------------------------------
// Fragment loader: 16 bf16 for one WMMA operand from a [row][K] bf16 buffer.
// Lane layout (CDNA5 16-bit A/B): elems 0..7 = K(khalf..khalf+7),
// elems 8..15 = K(16+khalf..16+khalf+7).  Two contiguous 16B loads.
// ---------------------------------------------------------------------------
__device__ __forceinline__ v16bf load_frag(const bf16* base) {
    v8bf lo = *(const v8bf*)(base);
    v8bf hi = *(const v8bf*)(base + 16);
    return __builtin_shufflevector(lo, hi, 0,1,2,3,4,5,6,7,8,9,10,11,12,13,14,15);
}

// ---------------------------------------------------------------------------
// Generic bf16 GEMM: C[M][N] = A[M][K] @ W^T (WT is [N][K]) + bias, opt tanh.
// Block = 256 threads (8 waves). Block tile 64(M) x 64(N); each wave 16x32.
// M % 64 == 0, N % 64 == 0, K % 32 == 0.
// ---------------------------------------------------------------------------
__global__ void gemm_bf16_wmma(const bf16* __restrict__ A,
                               const bf16* __restrict__ WT,
                               const float* __restrict__ bias,
                               float* __restrict__ C,
                               int M, int N, int K, int do_tanh) {
    const int lane = threadIdx.x & 31;
    const int wid  = threadIdx.x >> 5;
    const int wm   = wid & 3;          // 4 waves along M
    const int wn   = wid >> 2;         // 2 waves along N
    const int m0 = blockIdx.x * 64 + wm * 16;
    const int n0 = blockIdx.y * 64 + wn * 32;
    const int lm    = lane & 15;
    const int khalf = (lane >> 4) * 8;

    const bf16* ap  = A  + (size_t)(m0 + lm)       * K + khalf;
    const bf16* bp0 = WT + (size_t)(n0 + lm)       * K + khalf;
    const bf16* bp1 = WT + (size_t)(n0 + 16 + lm)  * K + khalf;

    v8f c0 = {}; v8f c1 = {};
    for (int k0 = 0; k0 < K; k0 += 32) {
        v16bf a  = load_frag(ap  + k0);
        v16bf b0 = load_frag(bp0 + k0);
        v16bf b1 = load_frag(bp1 + k0);
        c0 = __builtin_amdgcn_wmma_f32_16x16x32_bf16(false, a, false, b0,
                                                     (short)0, c0, false, false);
        c1 = __builtin_amdgcn_wmma_f32_16x16x32_bf16(false, a, false, b1,
                                                     (short)0, c1, false, false);
    }
    // C/D layout: vgpr i -> M = i (+8 for lanes 16..31); lane%16 -> N
    const int rbase = m0 + (lane >> 4) * 8;
    const int col0 = n0 + lm, col1 = n0 + 16 + lm;
    const float bi0 = bias[col0], bi1 = bias[col1];
    #pragma unroll
    for (int i = 0; i < 8; ++i) {
        const int r = rbase + i;
        float x0 = c0[i] + bi0;
        float x1 = c1[i] + bi1;
        if (do_tanh) { x0 = tanhf(x0); x1 = tanhf(x1); }
        C[(size_t)r * N + col0] = x0;
        C[(size_t)r * N + col1] = x1;
    }
}

// ---------------------------------------------------------------------------
// Weight transpose + f32->bf16 (+ zero pad K): W[K][N] -> WT[N][Kp]
// ---------------------------------------------------------------------------
__global__ void wtrans_kernel(const float* __restrict__ W, bf16* __restrict__ WT,
                              int K, int N, int Kp) {
    const int n = blockIdx.x;
    for (int k = threadIdx.x; k < Kp; k += blockDim.x)
        WT[(size_t)n * Kp + k] = (k < K) ? (bf16)W[(size_t)k * N + n] : (bf16)0.0f;
}

// gather word_table rows by id, pad DW->DWP, convert to bf16
__global__ void gather_pad_kernel(const float* __restrict__ table,
                                  const int* __restrict__ ids,
                                  bf16* __restrict__ gb) {
    const int r = blockIdx.x;                 // 0 .. B*W-1
    const int id = ids[r];
    const float* src = table + (size_t)id * DW_;
    bf16* dst = gb + (size_t)r * DWP;
    for (int k = threadIdx.x; k < DWP; k += blockDim.x)
        dst[k] = (k < DW_) ? (bf16)src[k] : (bf16)0.0f;
}

__global__ void f2bf_kernel(const float* __restrict__ in, bf16* __restrict__ out,
                            size_t n) {
    for (size_t i = blockIdx.x * (size_t)blockDim.x + threadIdx.x; i < n;
         i += (size_t)gridDim.x * blockDim.x)
        out[i] = (bf16)in[i];
}

__global__ void copy_f32_kernel(const float* __restrict__ in,
                                float* __restrict__ out, size_t n) {
    for (size_t i = blockIdx.x * (size_t)blockDim.x + threadIdx.x; i < n;
         i += (size_t)gridDim.x * blockDim.x)
        out[i] = in[i];
}

__global__ void zero_i32_kernel(int* __restrict__ p, int n) {
    int i = blockIdx.x * blockDim.x + threadIdx.x;
    if (i < n) p[i] = 0;
}

// ---------------------------------------------------------------------------
// Word mixing: one block per (b,w). Computes cosine softmax vs word embed,
// argmax (mic), x1 update, the mic/other mixing, writes x_mix rows and keep.
// ---------------------------------------------------------------------------
__global__ void word_mix_kernel(const float* __restrict__ x,
                                const float* __restrict__ we,
                                const int* __restrict__ widx,
                                const float* __restrict__ lam_p,
                                float* __restrict__ xm,
                                int* __restrict__ keep) {
    const int bw = blockIdx.x;          // b*W + w
    const int b  = bw / WW;
    const int start = widx[bw * 2 + 0];
    const int end   = widx[bw * 2 + 1];
    int n = end - start;
    if (n <= 0) return;                 // uniform across block
    if (n > LL) n = LL;

    const float* wer = we + (size_t)bw * HH;
    const float* xrow0 = x + ((size_t)b * SS) * HH;
    const int t = threadIdx.x;          // 256 threads

    __shared__ float red[256];

    float p_we2 = 0.f, p_dot[LL] = {0,0,0,0}, p_x2[LL] = {0,0,0,0};
    for (int d = t; d < HH; d += 256) {
        const float wv = wer[d];
        p_we2 += wv * wv;
        for (int l = 0; l < n; ++l) {
            const float xv = xrow0[(size_t)(start + l) * HH + d];
            p_dot[l] += xv * wv;
            p_x2[l]  += xv * xv;
        }
    }
    auto blk_reduce = [&](float v) -> float {
        red[t] = v; __syncthreads();
        for (int s = 128; s > 0; s >>= 1) {
            if (t < s) red[t] += red[t + s];
            __syncthreads();
        }
        const float r = red[0]; __syncthreads();
        return r;
    };
    const float we2 = blk_reduce(p_we2);
    float dotv[LL], x2v[LL];
    for (int l = 0; l < n; ++l) { dotv[l] = blk_reduce(p_dot[l]); x2v[l] = blk_reduce(p_x2[l]); }

    const float wnorm = sqrtf(we2);
    float cosv[LL];
    float cmax = -1e30f; int mic = 0;
    for (int l = 0; l < n; ++l) {
        cosv[l] = dotv[l] / (sqrtf(x2v[l]) * wnorm);
        if (cosv[l] > cmax) { cmax = cosv[l]; mic = l; }   // first-max like argmax
    }
    float sm[LL], esum = 0.f;
    for (int l = 0; l < n; ++l) { sm[l] = __expf(cosv[l] - cmax); esum += sm[l]; }
    for (int l = 0; l < n; ++l) sm[l] /= esum;

    const float lam = lam_p[0];
    const float f = __expf(lam - 1.0f);
    const float a = (n > 1) ? (1.0f - f) / (float)(n - 1) : 0.0f;

    for (int d = t; d < HH; d += 256) {
        const float wv = wer[d];
        float x1g[LL], sum_x = 0.f;
        for (int l = 0; l < n; ++l) {
            x1g[l] = xrow0[(size_t)(start + l) * HH + d] + sm[l] * wv;
            sum_x += x1g[l];
        }
        const float xmic = x1g[mic];
        for (int l = 0; l < n; ++l) {
            float nv;
            if (n == 1)        nv = x1g[l];
            else if (l == mic) nv = f * xmic + a * (sum_x - xmic);
            else               nv = (1.0f - a) * x1g[l] + a * xmic;
            xm[((size_t)b * SS + start + l) * HH + d] = nv;
        }
    }
    if (t == 0) {
        int kp = (n == 1) ? start : (start + mic);
        if (kp > SS - 1) kp = SS - 1;
        keep[b * SS + kp] = 1;
    }
}

// ---------------------------------------------------------------------------
// Q projection for token 0 only: q0[b][o] = x_mix[b,0,:] . Wq[:,o] + bq[o]
// grid (B, H/256), block 256.
// ---------------------------------------------------------------------------
__global__ void qproj_kernel(const float* __restrict__ xm,
                             const float* __restrict__ Wq,
                             const float* __restrict__ bq,
                             float* __restrict__ q0) {
    const int b = blockIdx.x;
    const int o = blockIdx.y * 256 + threadIdx.x;
    const float* xr = xm + (size_t)b * SS * HH;     // row 0
    float acc = bq[o];
    for (int k = 0; k < HH; ++k) acc += xr[k] * Wq[(size_t)k * HH + o];
    q0[(size_t)b * HH + o] = acc;
}

// ---------------------------------------------------------------------------
// Attention for query position 0. grid (B, NH), block 256.
// keep==nullptr -> no key mask (branch 1); else -inf where !keep (branch 2).
// ---------------------------------------------------------------------------
__global__ void attn0_kernel(const float* __restrict__ q0,
                             const float* __restrict__ Kmat,
                             const float* __restrict__ Vmat,
                             const int* __restrict__ keep,
                             float* __restrict__ out) {
    const int b = blockIdx.x, h = blockIdx.y;
    __shared__ float sc[SS];
    __shared__ float qd[HD];
    __shared__ float r2[256];
    const int t = threadIdx.x;

    if (t < HD) qd[t] = q0[(size_t)b * HH + h * HD + t];
    __syncthreads();

    for (int j = t; j < SS; j += 256) {
        const float* kr = Kmat + ((size_t)b * SS + j) * HH + h * HD;
        float s = 0.f;
        for (int d = 0; d < HD; ++d) s += qd[d] * kr[d];
        s *= 0.125f;                         // 1/sqrt(64)
        if (keep && !keep[b * SS + j]) s = -1e30f;
        sc[j] = s;
    }
    __syncthreads();

    float m = fmaxf(sc[t], sc[t + 256]);
    r2[t] = m; __syncthreads();
    for (int s = 128; s > 0; s >>= 1) { if (t < s) r2[t] = fmaxf(r2[t], r2[t + s]); __syncthreads(); }
    const float mx = r2[0]; __syncthreads();

    const float e0 = __expf(sc[t] - mx), e1 = __expf(sc[t + 256] - mx);
    sc[t] = e0; sc[t + 256] = e1;
    r2[t] = e0 + e1; __syncthreads();
    for (int s = 128; s > 0; s >>= 1) { if (t < s) r2[t] += r2[t + s]; __syncthreads(); }
    const float inv = 1.0f / r2[0]; __syncthreads();

    if (t < HD) {
        float acc = 0.f;
        for (int j = 0; j < SS; ++j)
            acc += sc[j] * Vmat[((size_t)b * SS + j) * HH + h * HD + t];
        out[(size_t)b * HH + h * HD + t] = acc * inv;
    }
}

// ---------------------------------------------------------------------------
// Head: fus/pooled/cls for token 0. grid B, block 256.
// ---------------------------------------------------------------------------
__global__ void head_kernel(const float* __restrict__ h1,
                            const float* __restrict__ h2,
                            const float* __restrict__ mu_p,
                            const float* __restrict__ fus_w, const float* __restrict__ fus_b,
                            const float* __restrict__ den_w, const float* __restrict__ den_b,
                            const float* __restrict__ cls_w, const float* __restrict__ cls_b,
                            float* __restrict__ out) {
    const int b = blockIdx.x;
    const int t = threadIdx.x;
    __shared__ float hm[HH];
    __shared__ float fus0[HH];
    __shared__ float pooled[HH];
    const float mu = mu_p[0];

    for (int d = t; d < HH; d += 256)
        hm[d] = mu * h1[(size_t)b * HH + d] + (1.0f - mu) * h2[(size_t)b * HH + d];
    __syncthreads();

    for (int o = t; o < HH; o += 256) {
        float acc = fus_b[o];
        for (int k = 0; k < HH; ++k) acc += hm[k] * fus_w[(size_t)k * HH + o];
        fus0[o] = tanhf(acc);
    }
    __syncthreads();

    for (int o = t; o < HH; o += 256) {
        float acc = den_b[o];
        for (int k = 0; k < HH; ++k) acc += fus0[k] * den_w[(size_t)k * HH + o];
        pooled[o] = tanhf(acc);
    }
    __syncthreads();

    if (t < NLL) {
        float acc = cls_b[t];
        for (int k = 0; k < HH; ++k) acc += pooled[k] * cls_w[(size_t)k * NLL + t];
        out[b * NLL + t] = acc;
    }
}

// ---------------------------------------------------------------------------
extern "C" void kernel_launch(void* const* d_in, const int* in_sizes, int n_in,
                              void* d_out, int out_size, void* d_ws, size_t ws_size,
                              hipStream_t stream) {
    // inputs (dict order)
    const float* x        = (const float*)d_in[0];   // (B,S,H)
    const int*   word_ids = (const int*)  d_in[1];   // (B,W)
    const int*   word_idx = (const int*)  d_in[2];   // (B,W,2)
    const float* table    = (const float*)d_in[3];   // (V+1,DW)
    const float* lam      = (const float*)d_in[4];
    const float* mu       = (const float*)d_in[5];
    const float* dt_w = (const float*)d_in[6];  const float* dt_b = (const float*)d_in[7];
    const float* dr_w = (const float*)d_in[8];  const float* dr_b = (const float*)d_in[9];
    const float* q_w  = (const float*)d_in[10]; const float* q_b  = (const float*)d_in[11];
    const float* k_w  = (const float*)d_in[12]; const float* k_b  = (const float*)d_in[13];
    const float* v_w  = (const float*)d_in[14]; const float* v_b  = (const float*)d_in[15];
    const float* qi_w = (const float*)d_in[16]; const float* qi_b = (const float*)d_in[17];
    const float* ki_w = (const float*)d_in[18]; const float* ki_b = (const float*)d_in[19];
    const float* vi_w = (const float*)d_in[20]; const float* vi_b = (const float*)d_in[21];
    const float* fus_w = (const float*)d_in[22]; const float* fus_b = (const float*)d_in[23];
    const float* den_w = (const float*)d_in[24]; const float* den_b = (const float*)d_in[25];
    const float* cls_w = (const float*)d_in[26]; const float* cls_b = (const float*)d_in[27];
    float* out = (float*)d_out;

    // workspace carve-up (all sizes are multiples of 256 bytes)
    char* ws = (char*)d_ws;
    size_t off = 0;
    auto alloc = [&](size_t bytes) { char* p = ws + off; off += (bytes + 255) & ~(size_t)255; return p; };
    const size_t MW = (size_t)BB * WW;   // 2048
    const size_t MS = (size_t)BB * SS;   // 4096

    bf16* gb    = (bf16*)alloc(MW * DWP * sizeof(bf16));
    bf16* dtT   = (bf16*)alloc((size_t)HH * DWP * sizeof(bf16));
    bf16* drT   = (bf16*)alloc((size_t)HH * HH * sizeof(bf16));
    bf16* kT    = (bf16*)alloc((size_t)HH * HH * sizeof(bf16));
    bf16* vT    = (bf16*)alloc((size_t)HH * HH * sizeof(bf16));
    bf16* kiT   = (bf16*)alloc((size_t)HH * HH * sizeof(bf16));
    bf16* viT   = (bf16*)alloc((size_t)HH * HH * sizeof(bf16));
    float* ehid = (float*)alloc(MW * HH * sizeof(float));
    bf16* ehidb = (bf16*)alloc(MW * HH * sizeof(bf16));
    float* we   = (float*)alloc(MW * HH * sizeof(float));
    float* xmix = (float*)alloc(MS * HH * sizeof(float));
    bf16* xb    = (bf16*)alloc(MS * HH * sizeof(bf16));
    float* k1   = (float*)alloc(MS * HH * sizeof(float));
    float* v1   = (float*)alloc(MS * HH * sizeof(float));
    float* k2   = (float*)alloc(MS * HH * sizeof(float));
    float* v2   = (float*)alloc(MS * HH * sizeof(float));
    int*   keep = (int*)  alloc(MS * sizeof(int));
    float* q1   = (float*)alloc((size_t)BB * HH * sizeof(float));
    float* q2   = (float*)alloc((size_t)BB * HH * sizeof(float));
    float* h1   = (float*)alloc((size_t)BB * HH * sizeof(float));
    float* h2   = (float*)alloc((size_t)BB * HH * sizeof(float));
    (void)ws_size; (void)in_sizes; (void)n_in; (void)out_size;

    // 1) weight prep (transpose -> [N][K] bf16, pad dt K to 320)
    wtrans_kernel<<<HH, 256, 0, stream>>>(dt_w, dtT, DW_, HH, DWP);
    wtrans_kernel<<<HH, 256, 0, stream>>>(dr_w, drT, HH, HH, HH);
    wtrans_kernel<<<HH, 256, 0, stream>>>(k_w,  kT,  HH, HH, HH);
    wtrans_kernel<<<HH, 256, 0, stream>>>(v_w,  vT,  HH, HH, HH);
    wtrans_kernel<<<HH, 256, 0, stream>>>(ki_w, kiT, HH, HH, HH);
    wtrans_kernel<<<HH, 256, 0, stream>>>(vi_w, viT, HH, HH, HH);

    // 2) word embedding path: gather -> GEMM(tanh) -> GEMM
    gather_pad_kernel<<<(int)MW, 256, 0, stream>>>(table, word_ids, gb);
    gemm_bf16_wmma<<<dim3(MW / 64, HH / 64), 256, 0, stream>>>(gb, dtT, dt_b, ehid,
                                                               (int)MW, HH, DWP, 1);
    f2bf_kernel<<<1024, 256, 0, stream>>>(ehid, ehidb, MW * HH);
    gemm_bf16_wmma<<<dim3(MW / 64, HH / 64), 256, 0, stream>>>(ehidb, drT, dr_b, we,
                                                               (int)MW, HH, HH, 0);

    // 3) word mixing -> x_mix, keep mask
    copy_f32_kernel<<<2048, 256, 0, stream>>>(x, xmix, MS * HH);
    zero_i32_kernel<<<(int)(MS + 255) / 256, 256, 0, stream>>>(keep, (int)MS);
    word_mix_kernel<<<(int)MW, 256, 0, stream>>>(x, we, word_idx, lam, xmix, keep);

    // 4) K/V projections via WMMA (both attention branches)
    f2bf_kernel<<<2048, 256, 0, stream>>>(xmix, xb, MS * HH);
    gemm_bf16_wmma<<<dim3(MS / 64, HH / 64), 256, 0, stream>>>(xb, kT,  k_b,  k1, (int)MS, HH, HH, 0);
    gemm_bf16_wmma<<<dim3(MS / 64, HH / 64), 256, 0, stream>>>(xb, vT,  v_b,  v1, (int)MS, HH, HH, 0);
    gemm_bf16_wmma<<<dim3(MS / 64, HH / 64), 256, 0, stream>>>(xb, kiT, ki_b, k2, (int)MS, HH, HH, 0);
    gemm_bf16_wmma<<<dim3(MS / 64, HH / 64), 256, 0, stream>>>(xb, viT, vi_b, v2, (int)MS, HH, HH, 0);

    // 5) Q at token 0 only (output depends only on fus[:,0])
    qproj_kernel<<<dim3(BB, HH / 256), 256, 0, stream>>>(xmix, q_w,  q_b,  q1);
    qproj_kernel<<<dim3(BB, HH / 256), 256, 0, stream>>>(xmix, qi_w, qi_b, q2);

    // 6) attention at query 0
    attn0_kernel<<<dim3(BB, NHH), 256, 0, stream>>>(q1, k1, v1, nullptr, h1);
    attn0_kernel<<<dim3(BB, NHH), 256, 0, stream>>>(q2, k2, v2, keep,    h2);

    // 7) fuse + pool + classify
    head_kernel<<<BB, 256, 0, stream>>>(h1, h2, mu, fus_w, fus_b, den_w, den_b,
                                        cls_w, cls_b, out);
}
